// CellPerformer_55843164782991
// MI455X (gfx1250) — compile-verified, matrix operand order checked by reading, not run on hardware
//
#include <hip/hip_runtime.h>
#include <hip/hip_bf16.h>
#include <math.h>

typedef _Float16 f16;
typedef __attribute__((ext_vector_type(8)))  _Float16 v8h;
typedef __attribute__((ext_vector_type(16))) _Float16 v16h;
typedef __attribute__((ext_vector_type(8)))  float    v8f;

#define Bb   4
#define Nn   2048
#define Mm   35
#define Dd   128
#define Hh   8
#define Ll   4
#define IDm  1024
#define FDv  621
#define FDP  640
#define FFv  512
#define Tt   (Bb * Nn)   /* 8192 tokens */
#define BH   (Bb * Hh)   /* 32 (b,h) slices */

#define NORMC   0.29730177875068026f   /* 128^-0.25                 */
#define HNORM2  0.04419417382415922f   /* 0.5 * 128^-0.5            */
#define RATIOC  0.04012862135913622f   /* 621^-0.5                  */
#define KEPS_C  1e-4f

// ---------------------------------------------------------------------------
// Generic wave-tile WMMA GEMM.  C[M,N] = A[M,K] * B[K,N] where BOTH operands
// are stored K-contiguous: A row-major [M,K] (lda), Bt = B^T row-major [N,K]
// (ldb).  Per the CDNA5 16-bit A-operand layout (lane m holds row m; halves
// 0..7 = K {lg*8..}, halves 8..15 = K {16+lg*8..}); the B operand is the
// A-layout of B^T, so one loader serves both.
// Wave tile: 16(M) x 64(N); block = 256 thr = 8 waves arranged 4(M) x 2(N)
// -> block tile 64 x 128.  All GEMMs here divide these tiles, and all K are
// multiples of 64, so the k-loop is manually unrolled x2 into a register
// ping-pong (compute buf0 / refill buf0 / compute buf1 / refill buf1): no
// rotation moves, and every load has 4-8 WMMAs of slack before its wait.
// ---------------------------------------------------------------------------
__device__ __forceinline__ v16h ld16(const f16* __restrict__ r) {
  v8h lo = *(const v8h*)(r);
  v8h hi = *(const v8h*)(r + 16);
  return __builtin_shufflevector(lo, hi, 0,1,2,3,4,5,6,7,8,9,10,11,12,13,14,15);
}

__device__ __forceinline__ float gelu_f(float x) {
  float x3 = x * x * x;
  return 0.5f * x * (1.0f + tanhf(0.7978845608028654f * (x + 0.044715f * x3)));
}

#define WMMA_F16(Aop, Bop, Cop)                                              \
  __builtin_amdgcn_wmma_f32_16x16x32_f16(false, (Aop), false, (Bop),         \
                                         (short)0, (Cop), false, false)

// MODE: 0 = f32 store *scale (xp)
//       1 = f16 head-scatter  q/k -> [B,H,N,128]   (+bias)
//       2 = f16 head-scatter  v   -> [B,H,128,N]   (+bias, transposed)
//       3 = f16 transposed store (ctx -> ctxT [128,FDP])
//       4 = f16 o-scatter * dinv -> [B,N,1024] (+h*128)
//       5 = f32 accumulate  y += c + bias
//       6 = f16 gelu(c + bias)
template <int MODE>
__global__ void __launch_bounds__(256)
wmma_gemm(const f16* __restrict__ A, int lda, long aB,
          const f16* __restrict__ Bt, int ldb, long bB,
          void* __restrict__ Out, int ldo, long oB,
          const float* __restrict__ bias,
          const float* __restrict__ aux, long auxB,
          int K, float scale) {
  const int z    = blockIdx.z;
  A  += (long)z * aB;
  Bt += (long)z * bB;
  const float* auxp = aux + (long)z * auxB;

  const int wave = threadIdx.x >> 5;
  const int lane = threadIdx.x & 31;
  const int lg   = lane >> 4;     // half-wave group (K interleave)
  const int lr   = lane & 15;     // row-in-tile (A) / col-in-tile (B,C)
  const int m0   = blockIdx.x * 64  + (wave >> 1) * 16;
  const int n0   = blockIdx.y * 128 + (wave & 1) * 64;

  v8f acc[4];
#pragma unroll
  for (int j = 0; j < 4; ++j)
#pragma unroll
    for (int i = 0; i < 8; ++i) acc[j][i] = 0.0f;

  // Per-lane operand row pointers (named scalars -> global addrspace
  // inference stays intact; bumped by 64 halves per unrolled iteration).
  const f16* pa  = A  + (long)(m0 + lr) * lda + lg * 8;
  const f16* pb0 = Bt + (long)(n0 +  0 + lr) * ldb + lg * 8;
  const f16* pb1 = Bt + (long)(n0 + 16 + lr) * ldb + lg * 8;
  const f16* pb2 = Bt + (long)(n0 + 32 + lr) * ldb + lg * 8;
  const f16* pb3 = Bt + (long)(n0 + 48 + lr) * ldb + lg * 8;

  // Prologue: fill both k-step buffers (k and k+32).
  v16h a0  = ld16(pa);       v16h a1  = ld16(pa  + 32);
  v16h b00 = ld16(pb0);      v16h b10 = ld16(pb0 + 32);
  v16h b01 = ld16(pb1);      v16h b11 = ld16(pb1 + 32);
  v16h b02 = ld16(pb2);      v16h b12 = ld16(pb2 + 32);
  v16h b03 = ld16(pb3);      v16h b13 = ld16(pb3 + 32);

  const int iters = (K >> 6) - 1;   // K is always a multiple of 64
  for (int it = 0; it < iters; ++it) {
    pa += 64; pb0 += 64; pb1 += 64; pb2 += 64; pb3 += 64;
    __builtin_prefetch(pa + 192, 0, 3);   // speculative, non-faulting
    // compute buffer 0, then refill it for the next iteration
    acc[0] = WMMA_F16(a0, b00, acc[0]);
    acc[1] = WMMA_F16(a0, b01, acc[1]);
    acc[2] = WMMA_F16(a0, b02, acc[2]);
    acc[3] = WMMA_F16(a0, b03, acc[3]);
    a0  = ld16(pa);
    b00 = ld16(pb0); b01 = ld16(pb1); b02 = ld16(pb2); b03 = ld16(pb3);
    // compute buffer 1, then refill it
    acc[0] = WMMA_F16(a1, b10, acc[0]);
    acc[1] = WMMA_F16(a1, b11, acc[1]);
    acc[2] = WMMA_F16(a1, b12, acc[2]);
    acc[3] = WMMA_F16(a1, b13, acc[3]);
    a1  = ld16(pa  + 32);
    b10 = ld16(pb0 + 32); b11 = ld16(pb1 + 32);
    b12 = ld16(pb2 + 32); b13 = ld16(pb3 + 32);
  }
  // Tail: consume both buffers.
  acc[0] = WMMA_F16(a0, b00, acc[0]);
  acc[1] = WMMA_F16(a0, b01, acc[1]);
  acc[2] = WMMA_F16(a0, b02, acc[2]);
  acc[3] = WMMA_F16(a0, b03, acc[3]);
  acc[0] = WMMA_F16(a1, b10, acc[0]);
  acc[1] = WMMA_F16(a1, b11, acc[1]);
  acc[2] = WMMA_F16(a1, b12, acc[2]);
  acc[3] = WMMA_F16(a1, b13, acc[3]);

  // C layout: VGPR r holds (row = r + 8*lg, col = lr) of the 16x16 tile.
  const int gmB = m0 + 8 * lg;
#pragma unroll
  for (int j = 0; j < 4; ++j) {
    const int gn = n0 + j * 16 + lr;
#pragma unroll
    for (int r = 0; r < 8; ++r) {
      const int gm = gmB + r;
      float c = acc[j][r];
      if (MODE == 0) {
        ((float*)Out)[(long)z * oB + (long)gm * ldo + gn] = c * scale;
      } else if (MODE == 1 || MODE == 2) {
        float v = c + bias[gn];
        int b2_ = gm >> 11, nn = gm & 2047, h = gn >> 7, d = gn & 127;
        long idx = (MODE == 1)
                       ? (((long)(b2_ * Hh + h) * Nn + nn) * Dd + d)
                       : (((long)(b2_ * Hh + h) * Dd + d) * Nn + nn);
        ((f16*)Out)[idx] = (f16)v;
      } else if (MODE == 3) {
        ((f16*)Out)[(long)z * oB + (long)gn * ldo + gm] = (f16)c;
      } else if (MODE == 4) {
        int b2_ = z >> 3, h = z & 7;
        ((f16*)Out)[((long)(b2_ * Nn + gm)) * IDm + h * Dd + gn] =
            (f16)(c * auxp[gm]);
      } else if (MODE == 5) {
        ((float*)Out)[(long)gm * ldo + gn] += c + bias[gn];
      } else if (MODE == 6) {
        ((f16*)Out)[(long)gm * ldo + gn] = (f16)gelu_f(c + bias[gn]);
      }
    }
  }
  (void)scale; (void)ldo;
}

// ---------------------------------------------------------------------------
// Support kernels (wave32-aware; one wave per token/row where applicable)
// ---------------------------------------------------------------------------
__global__ void enc_kernel(const float* __restrict__ x,
                           const float* __restrict__ w,
                           const float* __restrict__ b,
                           float* __restrict__ y) {
  long i = (long)blockIdx.x * 256 + threadIdx.x;      // over Tt*Dd
  int t = (int)(i >> 7), d = (int)(i & 127);
  const float* xr = x + (long)t * Mm;
  float s = b[d];
#pragma unroll
  for (int m = 0; m < Mm; ++m) s += xr[m] * w[m * Dd + d];
  y[i] = s;
}

__global__ void ln_kernel(const float* __restrict__ y,
                          const float* __restrict__ g,
                          const float* __restrict__ bt,
                          f16* __restrict__ out) {
  int t = blockIdx.x * 8 + (threadIdx.x >> 5);
  int lane = threadIdx.x & 31;
  const float* row = y + (long)t * Dd;
  float v[4], s = 0.f, s2 = 0.f;
#pragma unroll
  for (int i = 0; i < 4; ++i) {
    float x = row[lane * 4 + i];
    v[i] = x; s += x; s2 += x * x;
  }
#pragma unroll
  for (int off = 16; off; off >>= 1) {
    s  += __shfl_xor(s, off, 32);
    s2 += __shfl_xor(s2, off, 32);
  }
  float mu = s * (1.f / Dd);
  float rs = rsqrtf(s2 * (1.f / Dd) - mu * mu + 1e-5f);
  f16* o = out + (long)t * Dd;
#pragma unroll
  for (int i = 0; i < 4; ++i) {
    int d = lane * 4 + i;
    o[d] = (f16)((v[i] - mu) * rs * g[d] + bt[d]);
  }
}

__global__ void diag_kernel(const f16* __restrict__ q, float* __restrict__ dg) {
  int idx = blockIdx.x * 8 + (threadIdx.x >> 5);   // over BH*Nn tokens
  int lane = threadIdx.x & 31;
  const f16* row = q + (long)idx * Dd;
  float s = 0.f;
#pragma unroll
  for (int i = 0; i < 4; ++i) {
    float x = (float)row[lane * 4 + i];
    s += x * x;
  }
#pragma unroll
  for (int off = 16; off; off >>= 1) s += __shfl_xor(s, off, 32);
  if (lane == 0) dg[idx] = HNORM2 * s;
}

__global__ void rowmax_kernel(const float* __restrict__ xp, float* __restrict__ mx) {
  int idx = blockIdx.x * 8 + (threadIdx.x >> 5);   // over BH*Nn rows
  int lane = threadIdx.x & 31;
  const float* row = xp + (long)idx * FDP;
  float m = -INFINITY;
  for (int f = lane; f < FDv; f += 32) m = fmaxf(m, row[f]);  // real features only
#pragma unroll
  for (int off = 16; off; off >>= 1) m = fmaxf(m, __shfl_xor(m, off, 32));
  if (lane == 0) mx[idx] = m;
}

__global__ void bhmax_kernel(const float* __restrict__ mx, float* __restrict__ km) {
  int z = blockIdx.x, tid = threadIdx.x;
  float m = -INFINITY;
  for (int n = tid; n < Nn; n += 256) m = fmaxf(m, mx[(long)z * Nn + n]);
  __shared__ float sm[256];
  sm[tid] = m; __syncthreads();
  for (int s = 128; s; s >>= 1) {
    if (tid < s) sm[tid] = fmaxf(sm[tid], sm[tid + s]);
    __syncthreads();
  }
  if (tid == 0) km[z] = sm[0];
}

__global__ void qf_kernel(const float* __restrict__ xp, const float* __restrict__ dg,
                          const float* __restrict__ mx, f16* __restrict__ qf) {
  long i = (long)blockIdx.x * 256 + threadIdx.x;    // over BH*Nn*FDP
  int f = (int)(i % FDP);
  long row = i / FDP;
  float v = 0.f;
  if (f < FDv) v = RATIOC * (expf(xp[i] - dg[row] - mx[row]) + KEPS_C);
  qf[i] = (f16)v;
}

__global__ void kf_kernel(const float* __restrict__ xp, const float* __restrict__ dg,
                          const float* __restrict__ km, f16* __restrict__ kfT) {
  long i = (long)blockIdx.x * 256 + threadIdx.x;    // over BH*FDP*Nn (n fastest)
  int n  = (int)(i % Nn);
  long rf = i / Nn;
  int f  = (int)(rf % FDP);
  int z  = (int)(rf / FDP);
  long tok = (long)z * Nn + n;
  float v = 0.f;
  if (f < FDv) v = RATIOC * (expf(xp[tok * FDP + f] - dg[tok] - km[z]) + KEPS_C);
  kfT[i] = (f16)v;
}

__global__ void ksum_kernel(const f16* __restrict__ kfT, float* __restrict__ ks) {
  int idx = blockIdx.x * 8 + (threadIdx.x >> 5);   // over BH*FDP rows
  int lane = threadIdx.x & 31;
  const f16* row = kfT + (long)idx * Nn;
  float s = 0.f;
  for (int n = lane; n < Nn; n += 32) s += (float)row[n];
#pragma unroll
  for (int off = 16; off; off >>= 1) s += __shfl_xor(s, off, 32);
  if (lane == 0) ks[idx] = s;
}

__global__ void dinv_kernel(const f16* __restrict__ qf, const float* __restrict__ ks,
                            float* __restrict__ di) {
  int idx = blockIdx.x * 8 + (threadIdx.x >> 5);   // over BH*Nn tokens
  int lane = threadIdx.x & 31;
  int z = idx >> 11;
  const f16* row = qf + (long)idx * FDP;
  const float* k = ks + (long)z * FDP;
  float s = 0.f;
  for (int f = lane; f < FDP; f += 32) s += (float)row[f] * k[f];
#pragma unroll
  for (int off = 16; off; off >>= 1) s += __shfl_xor(s, off, 32);
  if (lane == 0) di[idx] = 1.0f / s;
}

__global__ void dec_kernel(const float* __restrict__ y, const float* __restrict__ w,
                           const float* __restrict__ b, float* __restrict__ out) {
  int t = blockIdx.x * 8 + (threadIdx.x >> 5);
  int lane = threadIdx.x & 31;
  const float* row = y + (long)t * Dd;
  float s = 0.f;
#pragma unroll
  for (int i = 0; i < 4; ++i) {
    int d = lane * 4 + i;
    s += row[d] * w[d];
  }
#pragma unroll
  for (int off = 16; off; off >>= 1) s += __shfl_xor(s, off, 32);
  if (lane == 0) out[t] = s + b[0];
}

__global__ void tconv_kernel(const float* __restrict__ src, f16* __restrict__ dst,
                             int R, int C) {          // src[R,C] -> dst[C,R] f16
  long i = (long)blockIdx.x * 256 + threadIdx.x;
  if (i >= (long)R * C) return;
  int r = (int)(i / C), c = (int)(i % C);
  dst[(long)c * R + r] = (f16)src[i];
}

__global__ void projconv_kernel(const float* __restrict__ src, f16* __restrict__ dst) {
  int i = blockIdx.x * 256 + threadIdx.x;            // FDP*128
  if (i >= FDP * Dd) return;
  int r = i >> 7;
  dst[i] = (r < FDv) ? (f16)src[i] : (f16)0.0f;
}

// ---------------------------------------------------------------------------
extern "C" void kernel_launch(void* const* d_in, const int* in_sizes, int n_in,
                              void* d_out, int out_size, void* d_ws, size_t ws_size,
                              hipStream_t stream) {
  const float* x     = (const float*)d_in[0];
  const float* enc_w = (const float*)d_in[1];
  const float* enc_b = (const float*)d_in[2];
  const float* ln1_g = (const float*)d_in[3];
  const float* ln1_b = (const float*)d_in[4];
  const float* wq    = (const float*)d_in[5];
  const float* bq    = (const float*)d_in[6];
  const float* wk    = (const float*)d_in[7];
  const float* bk    = (const float*)d_in[8];
  const float* wv    = (const float*)d_in[9];
  const float* bv    = (const float*)d_in[10];
  const float* wo    = (const float*)d_in[11];
  const float* bo    = (const float*)d_in[12];
  const float* ln2_g = (const float*)d_in[13];
  const float* ln2_b = (const float*)d_in[14];
  const float* w1    = (const float*)d_in[15];
  const float* b1    = (const float*)d_in[16];
  const float* w2    = (const float*)d_in[17];
  const float* b2    = (const float*)d_in[18];
  const float* proj  = (const float*)d_in[19];
  const float* dec_w = (const float*)d_in[20];
  const float* dec_b = (const float*)d_in[21];

  // -------- workspace carve (256B aligned) --------
  char* p = (char*)d_ws;
  auto alloc = [&](size_t bytes) -> char* {
    char* r = p;
    p += (bytes + 255) & ~(size_t)255;
    return r;
  };
  float* y      = (float*)alloc((size_t)Tt * Dd * 4);
  f16*   h16    = (f16*)  alloc((size_t)Tt * Dd * 2);
  f16*   q16    = (f16*)  alloc((size_t)BH * Nn * Dd * 2);
  f16*   k16    = (f16*)  alloc((size_t)BH * Nn * Dd * 2);
  f16*   vT16   = (f16*)  alloc((size_t)BH * Dd * Nn * 2);
  float* xp     = (float*)alloc((size_t)BH * Nn * FDP * 4);
  f16*   qf16   = (f16*)  alloc((size_t)BH * Nn * FDP * 2);
  f16*   kfT16  = (f16*)  alloc((size_t)BH * FDP * Nn * 2);
  f16*   ctxT16 = (f16*)  alloc((size_t)BH * Dd * FDP * 2);
  f16*   o16    = (f16*)  alloc((size_t)Tt * IDm * 2);
  f16*   ff16   = (f16*)  alloc((size_t)Tt * FFv * 2);
  float* ksum   = (float*)alloc((size_t)BH * FDP * 4);
  float* dinv   = (float*)alloc((size_t)BH * Nn * 4);
  float* diag   = (float*)alloc((size_t)BH * Nn * 4);
  float* mx     = (float*)alloc((size_t)BH * Nn * 4);
  float* kmax   = (float*)alloc((size_t)BH * 4);
  f16*   wqT    = (f16*)  alloc((size_t)Ll * IDm * Dd * 2);
  f16*   wkT    = (f16*)  alloc((size_t)Ll * IDm * Dd * 2);
  f16*   wvT    = (f16*)  alloc((size_t)Ll * IDm * Dd * 2);
  f16*   woT    = (f16*)  alloc((size_t)Ll * Dd * IDm * 2);
  f16*   w1T    = (f16*)  alloc((size_t)Ll * FFv * Dd * 2);
  f16*   w2T    = (f16*)  alloc((size_t)Ll * Dd * FFv * 2);
  f16*   proj16 = (f16*)  alloc((size_t)Ll * FDP * Dd * 2);

  // -------- weight prep (f32 -> f16, K-contiguous B^T layouts) --------
  for (int l = 0; l < Ll; ++l) {
    tconv_kernel<<<(Dd * IDm + 255) / 256, 256, 0, stream>>>(
        wq + (long)l * Dd * IDm, wqT + (long)l * IDm * Dd, Dd, IDm);
    tconv_kernel<<<(Dd * IDm + 255) / 256, 256, 0, stream>>>(
        wk + (long)l * Dd * IDm, wkT + (long)l * IDm * Dd, Dd, IDm);
    tconv_kernel<<<(Dd * IDm + 255) / 256, 256, 0, stream>>>(
        wv + (long)l * Dd * IDm, wvT + (long)l * IDm * Dd, Dd, IDm);
    tconv_kernel<<<(IDm * Dd + 255) / 256, 256, 0, stream>>>(
        wo + (long)l * IDm * Dd, woT + (long)l * Dd * IDm, IDm, Dd);
    tconv_kernel<<<(Dd * FFv + 255) / 256, 256, 0, stream>>>(
        w1 + (long)l * Dd * FFv, w1T + (long)l * FFv * Dd, Dd, FFv);
    tconv_kernel<<<(FFv * Dd + 255) / 256, 256, 0, stream>>>(
        w2 + (long)l * FFv * Dd, w2T + (long)l * Dd * FFv, FFv, Dd);
    projconv_kernel<<<(FDP * Dd + 255) / 256, 256, 0, stream>>>(
        proj + (long)l * FDv * Dd, proj16 + (long)l * FDP * Dd);
  }

  // -------- encoder --------
  enc_kernel<<<(Tt * Dd) / 256, 256, 0, stream>>>(x, enc_w, enc_b, y);

  for (int l = 0; l < Ll; ++l) {
    const f16* pj = proj16 + (long)l * FDP * Dd;

    // LN1 -> h16
    ln_kernel<<<Tt / 8, 256, 0, stream>>>(y, ln1_g + l * Dd, ln1_b + l * Dd, h16);

    // QKV projections (h16 [T,128] @ W^T[1024,128]) with head scatter
    {
      dim3 g(Tt / 64, IDm / 128, 1);
      wmma_gemm<1><<<g, 256, 0, stream>>>(h16, Dd, 0, wqT + (long)l * IDm * Dd, Dd, 0,
                                          q16, 0, 0, bq + (long)l * IDm, dinv, 0, Dd, 1.f);
      wmma_gemm<1><<<g, 256, 0, stream>>>(h16, Dd, 0, wkT + (long)l * IDm * Dd, Dd, 0,
                                          k16, 0, 0, bk + (long)l * IDm, dinv, 0, Dd, 1.f);
      wmma_gemm<2><<<g, 256, 0, stream>>>(h16, Dd, 0, wvT + (long)l * IDm * Dd, Dd, 0,
                                          vT16, 0, 0, bv + (long)l * IDm, dinv, 0, Dd, 1.f);
    }

    // ---- query FAVOR+ features ----
    diag_kernel<<<(BH * Nn) / 8, 256, 0, stream>>>(q16, diag);
    {
      dim3 g(Nn / 64, FDP / 128, BH);
      wmma_gemm<0><<<g, 256, 0, stream>>>(q16, Dd, (long)Nn * Dd, pj, Dd, 0,
                                          xp, FDP, (long)Nn * FDP,
                                          bq, dinv, 0, Dd, NORMC);
    }
    rowmax_kernel<<<(BH * Nn) / 8, 256, 0, stream>>>(xp, mx);
    qf_kernel<<<(int)(((long)BH * Nn * FDP + 255) / 256), 256, 0, stream>>>(xp, diag, mx, qf16);

    // ---- key FAVOR+ features (reuse xp/diag/mx buffers) ----
    diag_kernel<<<(BH * Nn) / 8, 256, 0, stream>>>(k16, diag);
    {
      dim3 g(Nn / 64, FDP / 128, BH);
      wmma_gemm<0><<<g, 256, 0, stream>>>(k16, Dd, (long)Nn * Dd, pj, Dd, 0,
                                          xp, FDP, (long)Nn * FDP,
                                          bk, dinv, 0, Dd, NORMC);
    }
    rowmax_kernel<<<(BH * Nn) / 8, 256, 0, stream>>>(xp, mx);
    bhmax_kernel<<<BH, 256, 0, stream>>>(mx, kmax);
    kf_kernel<<<(int)(((long)BH * FDP * Nn + 255) / 256), 256, 0, stream>>>(xp, diag, kmax, kfT16);

    // ksum[f] = sum_n kf ;  dinv[n] = 1 / (qf . ksum)
    ksum_kernel<<<(BH * FDP) / 8, 256, 0, stream>>>(kfT16, ksum);
    dinv_kernel<<<(BH * Nn) / 8, 256, 0, stream>>>(qf16, ksum, dinv);

    // ctx^T[d,f] = (kf^T @ v)[f,d] : A=kfT [FDP,Nn], Bt=vT [128,Nn], K=Nn
    {
      dim3 g(FDP / 64, Dd / 128, BH);
      wmma_gemm<3><<<g, 256, 0, stream>>>(kfT16, Nn, (long)FDP * Nn,
                                          vT16, Nn, (long)Dd * Nn,
                                          ctxT16, FDP, (long)Dd * FDP,
                                          bo, dinv, 0, Nn, 1.f);
    }

    // o = (qf @ ctx) * dinv  -> o16 [B,N,1024] : A=qf [Nn,FDP], Bt=ctxT [128,FDP]
    {
      dim3 g(Nn / 64, Dd / 128, BH);
      wmma_gemm<4><<<g, 256, 0, stream>>>(qf16, FDP, (long)Nn * FDP,
                                          ctxT16, FDP, (long)Dd * FDP,
                                          o16, 0, 0, bo, dinv, Nn, FDP, 1.f);
    }

    // y += o @ wo + bo
    {
      dim3 g(Tt / 64, Dd / 128, 1);
      wmma_gemm<5><<<g, 256, 0, stream>>>(o16, IDm, 0, woT + (long)l * Dd * IDm, IDm, 0,
                                          y, Dd, 0, bo + (long)l * Dd, dinv, 0, IDm, 1.f);
    }

    // FFN: y += gelu(LN2(y) @ w1 + b1) @ w2 + b2
    ln_kernel<<<Tt / 8, 256, 0, stream>>>(y, ln2_g + l * Dd, ln2_b + l * Dd, h16);
    {
      dim3 g(Tt / 64, FFv / 128, 1);
      wmma_gemm<6><<<g, 256, 0, stream>>>(h16, Dd, 0, w1T + (long)l * FFv * Dd, Dd, 0,
                                          ff16, FFv, 0, b1 + (long)l * FFv, dinv, 0, Dd, 1.f);
    }
    {
      dim3 g(Tt / 64, Dd / 128, 1);
      wmma_gemm<5><<<g, 256, 0, stream>>>(ff16, FFv, 0, w2T + (long)l * Dd * FFv, FFv, 0,
                                          y, Dd, 0, b2 + (long)l * Dd, dinv, 0, FFv, 1.f);
    }
  }

  // -------- decoder --------
  dec_kernel<<<Tt / 8, 256, 0, stream>>>(y, dec_w, dec_b, (float*)d_out);

  (void)in_sizes; (void)n_in; (void)out_size; (void)ws_size;
}
